// RansLinear_44598940402131
// MI455X (gfx1250) — compile-verified, matrix operand order checked by reading, not run on hardware
//
#include <hip/hip_runtime.h>
#include <hip/hip_bf16.h>
#include <stdint.h>
#include <stddef.h>

// CDNA5 wave32 WMMA fragment types (probe-confirmed builtin signatures).
typedef __attribute__((ext_vector_type(16))) __bf16 v16bf;
typedef __attribute__((ext_vector_type(8)))  float  v8f;
// Tensor-DMA descriptor groups (amdgpu-toolchain clang-23 6-arg builtin form).
typedef __attribute__((ext_vector_type(4))) unsigned int u32x4;
typedef __attribute__((ext_vector_type(8))) int          i32x8;
typedef __attribute__((ext_vector_type(4))) int          i32x4;

namespace {
constexpr int K_IN  = 4096;
constexpr int N_OUT = 4096;
constexpr int BM = 128;   // block tile M
constexpr int BN = 128;   // block tile N
constexpr int BK = 32;    // K slab = one v_wmma_f32_16x16x32_bf16 step
// LDS row pitch in dwords: 16 data dwords (32 bf16) + 4 pad dwords.
// 20 dwords = 80 B (16B aligned rows); the TDM pad feature (interval=16dw,
// amount=4dw) reproduces exactly this pitch when DMA-ing the A tile.
constexpr int LDP = 20;
}

union FragU {
  unsigned int u[8];
  v16bf v;
};

__device__ __forceinline__ unsigned bf16bits(unsigned e, unsigned m) {
  // bf16 = [sign:1][exp:8][man:7]; m carries sign in bit7, mantissa in bits0-6.
  return ((m & 0x80u) << 8) | ((e & 0xFFu) << 7) | (m & 0x7Fu);
}

__global__ __launch_bounds__(256)
void rans_decode_gemm_bf16(const __hip_bfloat16* __restrict__ X,
                           const int* __restrict__ EB,   // exponent bytes [K,N] as int32
                           const int* __restrict__ MB,   // sign+mantissa bytes [K,N] as int32
                           __hip_bfloat16* __restrict__ Y,
                           int M)
{
  __shared__ unsigned int As[BM * LDP];  // x tile (TDM-filled), bf16 pairs per dword
  __shared__ unsigned int Bs[BN * LDP];  // decoded W tile, per-column K-pair dwords

  const int tid  = threadIdx.x;
  const int lane = tid & 31;
  const int wid  = tid >> 5;   // 8 waves
  const int wm   = wid & 3;    // wave row: 4 x 32 rows
  const int wn   = wid >> 2;   // wave col: 2 x 64 cols
  const int half = lane >> 4;  // ISA 16-bit operand layout: lanes 0-15 / 16-31
  const int l16  = lane & 15;

  const int bm = blockIdx.y * BM;
  const int bn = blockIdx.x * BN;

  // B decode split: each thread decodes 4 consecutive n x 4 consecutive k.
  const int nb = (tid & 31) * 4;   // n base within tile (0..124)
  const int kg = tid >> 5;         // k group: k = 4*kg .. 4*kg+3 (kg = 0..7)

  // TDM descriptor pieces that are loop-invariant (wave 0 issues the DMA).
  const unsigned lds_a = (unsigned)(size_t)(void*)As;  // flat->LDS offset (low 32 bits)

  v8f zero = {};
  v8f acc[2][4];
#pragma unroll
  for (int i = 0; i < 2; ++i)
#pragma unroll
    for (int j = 0; j < 4; ++j) acc[i][j] = zero;

  for (int kt = 0; kt < K_IN; kt += BK) {
    // ---- stage A via the Tensor Data Mover: one descriptor DMAs the whole
    //      128x32 bf16 tile global->LDS with the 20-dword padded pitch ----
    if (tid < 32) {
      const unsigned long long ga =
          (unsigned long long)(size_t)(const void*)(X + (size_t)bm * K_IN + kt);
      const u32x4 g0 = {
          1u,                                                   // count=1 (valid user D#)
          lds_a,                                                // lds_addr
          (unsigned)ga,                                         // global_addr[31:0]
          ((unsigned)(ga >> 32) & 0x01FFFFFFu) | (2u << 30)     // global_addr[56:32] | type=2
      };
      const i32x8 g1 = {
          (int)((1u << 16) | (1u << 20) | (3u << 22) | (3u << 25)),
          //    data_size=2B | pad_en   | interval=16dw | amount=4dw
          (int)(((unsigned)K_IN & 0xFFFFu) << 16),              // tensor_dim0 lo16
          (int)((((unsigned)K_IN >> 16) & 0xFFFFu) |
                (((unsigned)M & 0xFFFFu) << 16)),               // dim0 hi16 | tensor_dim1 lo16
          (int)((((unsigned)M >> 16) & 0xFFFFu) | (32u << 16)), // dim1 hi16 | tile_dim0=32
          (int)128,                                             // tile_dim1=128, tile_dim2=0
          (int)K_IN,                                            // tensor_dim0_stride lo32
          0, 0                                                  // stride hi, dim1_stride (2D)
      };
      const i32x4 gz4 = {0, 0, 0, 0};
      const i32x8 gz8 = {0, 0, 0, 0, 0, 0, 0, 0};
      __builtin_amdgcn_tensor_load_to_lds(g0, g1, gz4, gz4, gz8, 0);
    }

    // ---- decode B while the TDM runs: 32 x 128 weight slab ----
#pragma unroll
    for (int j = 0; j < 2; ++j) {
      // two k-pairs per j-step: k = 4*kg + {2j, 2j+1}
      const size_t gidx0 = (size_t)(kt + 4 * kg + 2 * j) * N_OUT + (size_t)(bn + nb);
      const int4 e0 = *reinterpret_cast<const int4*>(EB + gidx0);
      const int4 m0 = *reinterpret_cast<const int4*>(MB + gidx0);
      const int4 e1 = *reinterpret_cast<const int4*>(EB + gidx0 + N_OUT);
      const int4 m1 = *reinterpret_cast<const int4*>(MB + gidx0 + N_OUT);
      const unsigned lo[4] = { bf16bits(e0.x, m0.x), bf16bits(e0.y, m0.y),
                               bf16bits(e0.z, m0.z), bf16bits(e0.w, m0.w) };
      const unsigned hi[4] = { bf16bits(e1.x, m1.x), bf16bits(e1.y, m1.y),
                               bf16bits(e1.z, m1.z), bf16bits(e1.w, m1.w) };
#pragma unroll
      for (int i = 0; i < 4; ++i)
        Bs[(nb + i) * LDP + 2 * kg + j] = lo[i] | (hi[i] << 16);
    }

    // Prefetch next slab's weight bytes (global_prefetch_b8).
    if (kt + BK < K_IN) {
      const size_t gpf = (size_t)(kt + BK + 4 * kg) * N_OUT + (size_t)(bn + nb);
      __builtin_prefetch(EB + gpf, 0, 0);
      __builtin_prefetch(MB + gpf, 0, 0);
    }

    // Wave 0 waits for its tensor DMA, then everyone syncs (DScnt handled
    // by the compiler's barrier lowering).
    if (tid < 32) __builtin_amdgcn_s_wait_tensorcnt(0);
    __syncthreads();

    // ---- fragment loads per the 16-bit A/B VGPR layout (ISA 7.12.2) ----
    FragU afr[2];
#pragma unroll
    for (int mt = 0; mt < 2; ++mt) {
      const unsigned* Ar = &As[(wm * 32 + mt * 16 + l16) * LDP];
#pragma unroll
      for (int v = 0; v < 4; ++v) {
        afr[mt].u[v]     = Ar[half * 4 + v];
        afr[mt].u[4 + v] = Ar[8 + half * 4 + v];
      }
    }
    FragU bfr[4];
#pragma unroll
    for (int nt = 0; nt < 4; ++nt) {
      const unsigned* Br = &Bs[(wn * 64 + nt * 16 + l16) * LDP];
#pragma unroll
      for (int v = 0; v < 4; ++v) {
        bfr[nt].u[v]     = Br[half * 4 + v];
        bfr[nt].u[4 + v] = Br[8 + half * 4 + v];
      }
    }

    // ---- 8 WMMAs: 2 (M) x 4 (N) 16x16 tiles, K=32 ----
#pragma unroll
    for (int mt = 0; mt < 2; ++mt)
#pragma unroll
      for (int nt = 0; nt < 4; ++nt)
        acc[mt][nt] = __builtin_amdgcn_wmma_f32_16x16x32_bf16(
            /*neg_a=*/false, afr[mt].v,
            /*neg_b=*/false, bfr[nt].v,
            /*c_mod=*/(short)0, acc[mt][nt],
            /*reuse_a=*/false, /*reuse_b=*/false);

    __syncthreads();
  }

  // ---- epilogue: f32 accum -> bf16, per the C/D layout (VGPR v -> M = v + 8*half) ----
#pragma unroll
  for (int mt = 0; mt < 2; ++mt)
#pragma unroll
    for (int nt = 0; nt < 4; ++nt) {
      const int col = bn + wn * 64 + nt * 16 + l16;
#pragma unroll
      for (int v = 0; v < 8; ++v) {
        const int row = bm + wm * 32 + mt * 16 + half * 8 + v;
        Y[(size_t)row * N_OUT + col] = __float2bfloat16(acc[mt][nt][v]);
      }
    }
}

extern "C" void kernel_launch(void* const* d_in, const int* in_sizes, int n_in,
                              void* d_out, int out_size, void* d_ws, size_t ws_size,
                              hipStream_t stream) {
  (void)n_in; (void)out_size; (void)d_ws; (void)ws_size;
  const __hip_bfloat16* x  = (const __hip_bfloat16*)d_in[0];
  const int*            eb = (const int*)d_in[1];
  const int*            mb = (const int*)d_in[2];
  __hip_bfloat16*       y  = (__hip_bfloat16*)d_out;

  const int m_tokens = in_sizes[0] / K_IN;   // 8192
  dim3 grid(N_OUT / BN, m_tokens / BM);      // (32, 64)
  dim3 block(256);
  rans_decode_gemm_bf16<<<grid, block, 0, stream>>>(x, eb, mb, y, m_tokens);
}
